// ScanBranch_78752520340025
// MI455X (gfx1250) — compile-verified, matrix operand order checked by reading, not run on hardware
//
#include <hip/hip_runtime.h>
#include <hip/hip_bf16.h>
#include <math.h>

// ---------------------------------------------------------------------------
// MI455X (gfx1250, wave32) implementation.
// Heavy math (encoder projections, FF, self-attention) runs on
// v_wmma_f32_16x16x32_bf16 with explicit double-buffered fragment loads so
// loadcnt waits overlap the WMMA chain. Small glue (KAN, LN, BN, decoder
// Nq=1, tail) is scalar VALU. Workspace need ~166 MB (see offsets below).
// ---------------------------------------------------------------------------

#define T_TOK 16384      // B*N = 8*2048
#define NSEQ  2048
#define NB    8
#define DHID  512
#define EPSF  1e-5f

typedef __attribute__((ext_vector_type(16))) __bf16 v16bf;
typedef __attribute__((ext_vector_type(8)))  float  v8f;

struct alignas(16) U128 { unsigned int a, b, c, d; };
union FragB { v16bf v; U128 q[2]; unsigned short s[16]; };

__device__ __forceinline__ unsigned short f2bfu(float f) {
  unsigned int u = __float_as_uint(f);
  unsigned int r = u + 0x7FFFu + ((u >> 16) & 1u);
  return (unsigned short)(r >> 16);
}
__device__ __forceinline__ float bfu2f(unsigned short s) {
  return __uint_as_float(((unsigned int)s) << 16);
}
// 16-bit WMMA operand element->k mapping (ISA 7.12.2, 16-bit A 16x32):
// lane half (L>>4) selects +8; e in [0,8) -> k=e, e in [8,16) -> k=16+(e-8).
__device__ __forceinline__ int kmap(int e, int hf) {
  return ((e >> 3) << 4) + (e & 7) + (hf << 3);
}
__device__ __forceinline__ v8f zero8() {
  v8f z;
#pragma unroll
  for (int i = 0; i < 8; ++i) z[i] = 0.f;
  return z;
}
__device__ __forceinline__ v8f wmma_bf16(const FragB& a, const FragB& b, v8f c) {
  return __builtin_amdgcn_wmma_f32_16x16x32_bf16(false, a.v, false, b.v,
                                                 (short)0, c, false, false);
}
__device__ __forceinline__ void ld2x16(FragB& f, const unsigned short* p0,
                                       const unsigned short* p1) {
  f.q[0] = *reinterpret_cast<const U128*>(p0);
  f.q[1] = *reinterpret_cast<const U128*>(p1);
}

// ---------------------------------------------------------------------------
// KAN front-end: 4 KANLinear layers per token + LayerNorm(5)
// ---------------------------------------------------------------------------
template <int FI, int FO>
__device__ __forceinline__ void kan_layer(const float* __restrict__ bw,
                                          const float* __restrict__ sw,
                                          const float* __restrict__ gr,
                                          const float* hin, float* hout) {
#pragma unroll
  for (int o = 0; o < FO; ++o) hout[o] = 0.f;
  for (int i = 0; i < FI; ++i) {
    float xv = hin[i];
    const float* g = gr + i * 12;
    float bb[11];
#pragma unroll
    for (int j = 0; j < 11; ++j)
      bb[j] = (xv >= g[j] && xv < g[j + 1]) ? 1.f : 0.f;
#pragma unroll
    for (int k = 1; k <= 3; ++k)
      for (int j = 0; j + k < 11; ++j)
        bb[j] = (xv - g[j]) / (g[j + k] - g[j]) * bb[j]
              + (g[j + k + 1] - xv) / (g[j + k + 1] - g[j + 1]) * bb[j + 1];
    float sil = xv / (1.f + __expf(-xv));
    for (int o = 0; o < FO; ++o) {
      const float* swp = sw + ((size_t)o * FI + i) * 8;
      float acc = sil * bw[o * FI + i];
#pragma unroll
      for (int j = 0; j < 8; ++j) acc += swp[j] * bb[j];
      hout[o] += acc;
    }
  }
}

__global__ __launch_bounds__(256) void kan_kernel(
    const float* __restrict__ x,
    const float* bw0, const float* sw0, const float* g0,
    const float* bw1, const float* sw1, const float* g1,
    const float* bw2, const float* sw2, const float* g2,
    const float* bw3, const float* sw3, const float* g3,
    const float* lng, const float* lnb, float* __restrict__ h5) {
  int t = blockIdx.x * 256 + threadIdx.x;
  if (t >= T_TOK) return;
  int b = t / NSEQ, nn = t % NSEQ;
  float ha[16], hb[16];
#pragma unroll
  for (int c = 0; c < 5; ++c) ha[c] = x[((size_t)b * 5 + c) * NSEQ + nn];
  kan_layer<5, 8>(bw0, sw0, g0, ha, hb);
  kan_layer<8, 16>(bw1, sw1, g1, hb, ha);
  kan_layer<16, 8>(bw2, sw2, g2, ha, hb);
  kan_layer<8, 5>(bw3, sw3, g3, hb, ha);
  float m = 0.f;
#pragma unroll
  for (int c = 0; c < 5; ++c) m += ha[c];
  m *= 0.2f;
  float v = 0.f;
#pragma unroll
  for (int c = 0; c < 5; ++c) { float d = ha[c] - m; v += d * d; }
  float rstd = rsqrtf(v * 0.2f + EPSF);
#pragma unroll
  for (int c = 0; c < 5; ++c)
    h5[(size_t)t * 5 + c] = (ha[c] - m) * rstd * lng[c] + lnb[c];
}

__global__ void zero_stats(float* stats) {
  if (threadIdx.x < 16) stats[threadIdx.x] = 0.f;
}

__global__ __launch_bounds__(256) void bn5_stats(const float* __restrict__ h5,
                                                 float* __restrict__ stats) {
  __shared__ float bs[10];
  int tid = threadIdx.x;
  if (tid < 10) bs[tid] = 0.f;
  __syncthreads();
  float s[5] = {0, 0, 0, 0, 0}, ss[5] = {0, 0, 0, 0, 0};
  for (int t = blockIdx.x * 256 + tid; t < T_TOK; t += gridDim.x * 256) {
#pragma unroll
    for (int c = 0; c < 5; ++c) {
      float v = h5[(size_t)t * 5 + c];
      s[c] += v; ss[c] += v * v;
    }
  }
#pragma unroll
  for (int c = 0; c < 5; ++c) {
    atomicAdd(&bs[c], s[c]);
    atomicAdd(&bs[5 + c], ss[c]);
  }
  __syncthreads();
  if (tid < 10) atomicAdd(&stats[tid], bs[tid]);
}

// BN(5) + embedding 5->512, writes f32 X and bf16 Xb
__global__ __launch_bounds__(256) void embed_kernel(
    const float* __restrict__ h5, const float* __restrict__ stats,
    const float* __restrict__ bng, const float* __restrict__ bnb,
    const float* __restrict__ ew, const float* __restrict__ eb,
    float* __restrict__ X, unsigned short* __restrict__ Xb) {
  size_t idx = (size_t)blockIdx.x * 256 + threadIdx.x;
  if (idx >= (size_t)T_TOK * DHID) return;
  int t = (int)(idx >> 9), n = (int)(idx & 511);
  float val = eb[n];
#pragma unroll
  for (int c = 0; c < 5; ++c) {
    float mean = stats[c] * (1.f / T_TOK);
    float var = stats[5 + c] * (1.f / T_TOK) - mean * mean;
    float hb = (h5[(size_t)t * 5 + c] - mean) * rsqrtf(var + EPSF) * bng[c] + bnb[c];
    val += hb * ew[n * 5 + c];
  }
  X[idx] = val;
  Xb[idx] = f2bfu(val);
}

// ---------------------------------------------------------------------------
// Weight packing: f32 W[512,512] (row-major out,in) -> bf16 B-fragment order
// layout: [ntile(32)][ktile(16)][lane(32)][e(16)]
// ---------------------------------------------------------------------------
__global__ __launch_bounds__(256) void pack_w(const float* __restrict__ W,
                                              unsigned short* __restrict__ Wp) {
  int t = blockIdx.x * 256 + threadIdx.x;
  int lane = t & 31, f = t >> 5;        // f = nt*16 + kt, 512 frags
  int kt = f & 15, nt = f >> 4;
  int l16 = lane & 15, hf = lane >> 4;
  unsigned short* out = Wp + ((size_t)f * 32 + lane) * 16;
  const float* wrow = W + (size_t)(nt * 16 + l16) * 512 + kt * 32;
#pragma unroll
  for (int e = 0; e < 16; ++e) out[e] = f2bfu(wrow[kmap(e, hf)]);
}

// K pack: kb[T,512] bf16 -> per (b,h) B-fragments of K^T (chan x key)
// layout: [bh(32)][jtile(128)][kstep(4)][lane(32)][e(16)]
__global__ __launch_bounds__(256) void pack_k(const unsigned short* __restrict__ kb,
                                              unsigned short* __restrict__ kp) {
  int t = blockIdx.x * 256 + threadIdx.x;
  int lane = t & 31, rest = t >> 5;
  int ks = rest & 3; rest >>= 2;
  int jt = rest & 127; int bh = rest >> 7;
  int b = bh >> 2, h = bh & 3;
  int l16 = lane & 15, hf = lane >> 4;
  unsigned short* out = kp + ((size_t)(((bh * 128 + jt) * 4 + ks) * 32 + lane)) * 16;
  const unsigned short* src =
      kb + (size_t)(b * NSEQ + jt * 16 + l16) * 512 + h * 128 + ks * 32;
#pragma unroll
  for (int e = 0; e < 16; ++e) out[e] = src[kmap(e, hf)];
}

// V pack: vb[T,512] bf16 -> B-fragments of V (key x chan)
// layout: [bh(32)][kchunk(64)][ntile(8)][lane(32)][e(16)]
__global__ __launch_bounds__(256) void pack_v(const unsigned short* __restrict__ vb,
                                              unsigned short* __restrict__ vp) {
  int t = blockIdx.x * 256 + threadIdx.x;
  int lane = t & 31, rest = t >> 5;
  int nt = rest & 7; rest >>= 3;
  int kc = rest & 63; int bh = rest >> 6;
  int b = bh >> 2, h = bh & 3;
  int l16 = lane & 15, hf = lane >> 4;
  unsigned short* out = vp + ((size_t)(((bh * 64 + kc) * 8 + nt) * 32 + lane)) * 16;
  int chan = h * 128 + nt * 16 + l16;
#pragma unroll
  for (int e = 0; e < 16; ++e) {
    int key = kc * 32 + kmap(e, hf);
    out[e] = vb[(size_t)(b * NSEQ + key) * 512 + chan];
  }
}

// ---------------------------------------------------------------------------
// WMMA GEMM: C[M,512] = A[M,512](bf16) x W^T + bias, M = T_TOK
// block 256 = 8 waves, each wave 16(M) x 64(N); grid (M/128, 8).
// Fully unrolled K loop with double-buffered A/B fragments: iteration kt+1's
// loads issue before iteration kt's WMMAs, so loadcnt waits retire partially.
// mode 0: store f32 (outF, opt) and bf16 (outB, opt)
// mode 1: resX += relu(acc + bias)    (fused FF residual)
// ---------------------------------------------------------------------------
__global__ __launch_bounds__(256) void gemm512(
    const unsigned short* __restrict__ Ab, const unsigned short* __restrict__ Wp,
    const float* __restrict__ bias, float* __restrict__ outF,
    unsigned short* __restrict__ outB, float* __restrict__ resX, int mode) {
  int wave = threadIdx.x >> 5, lane = threadIdx.x & 31;
  int l16 = lane & 15, hf = lane >> 4;
  int m0 = blockIdx.x * 128 + wave * 16;
  int nb = blockIdx.y;  // 0..7
  v8f acc[4];
#pragma unroll
  for (int j = 0; j < 4; ++j) acc[j] = zero8();
  const unsigned short* arow = Ab + (size_t)(m0 + l16) * 512 + hf * 8;
  const unsigned short* wl = Wp + (size_t)lane * 16;  // + frag_id*512

  FragB af[2], bf[2][4];
  auto loadA = [&](int kt, FragB& f) {
    ld2x16(f, arow + kt * 32, arow + kt * 32 + 16);
  };
  auto loadB = [&](int kt, FragB* f) {
#pragma unroll
    for (int j = 0; j < 4; ++j) {
      const unsigned short* p = wl + ((size_t)((nb * 4 + j) * 16 + kt)) * 512;
      ld2x16(f[j], p, p + 8);
    }
  };
  loadA(0, af[0]);
  loadB(0, bf[0]);
#pragma unroll
  for (int kt = 0; kt < 16; ++kt) {
    int cur = kt & 1, nxt = cur ^ 1;
    if (kt < 15) {            // issue next-step loads before this step's math
      loadA(kt + 1, af[nxt]);
      loadB(kt + 1, bf[nxt]);
    }
#pragma unroll
    for (int j = 0; j < 4; ++j) acc[j] = wmma_bf16(af[cur], bf[cur][j], acc[j]);
  }
#pragma unroll
  for (int j = 0; j < 4; ++j) {
    int n = nb * 64 + j * 16 + l16;
    float bi = bias[n];
#pragma unroll
    for (int r = 0; r < 8; ++r) {
      int m = m0 + r + hf * 8;
      size_t idx = (size_t)m * 512 + n;
      float val = acc[j][r] + bi;
      if (mode == 1) {
        resX[idx] += fmaxf(val, 0.f);
      } else {
        if (outF) outF[idx] = val;
        if (outB) outB[idx] = f2bfu(val);
      }
    }
  }
}

// ---------------------------------------------------------------------------
// Flash self-attention (encoder): per wave 16 q-rows x dh=128, online softmax
// over 64 chunks of 32 keys. O = qh + softmax(QK^T/sqrt(512)) V -> Xout (f32)
// ---------------------------------------------------------------------------
__global__ __launch_bounds__(128) void flash_attn(
    const unsigned short* __restrict__ qb, const float* __restrict__ qf,
    const unsigned short* __restrict__ kp, const unsigned short* __restrict__ vp,
    float* __restrict__ Xout) {
  const int wave = threadIdx.x >> 5, lane = threadIdx.x & 31;
  const int l16 = lane & 15, hf = lane >> 4;
  const int bh = blockIdx.y, b = bh >> 2, h = bh & 3;
  const int q0 = blockIdx.x * 64 + wave * 16;
  const float scale = 0.044194173824159216f;  // 1/sqrt(512)

  __shared__ unsigned short Pt[4][16 * 40];
  unsigned short* P = &Pt[wave][0];

  FragB Qf[4];
  {
    const unsigned short* qrow =
        qb + (size_t)(b * NSEQ + q0 + l16) * 512 + h * 128 + hf * 8;
#pragma unroll
    for (int ks = 0; ks < 4; ++ks)
      ld2x16(Qf[ks], qrow + ks * 32, qrow + ks * 32 + 16);
  }
  v8f Oacc[8];
#pragma unroll
  for (int nt = 0; nt < 8; ++nt) Oacc[nt] = zero8();
  float mrow[8], lrow[8], corr[8];
#pragma unroll
  for (int r = 0; r < 8; ++r) { mrow[r] = -1e30f; lrow[r] = 0.f; }

  const unsigned short* kbase = kp + (size_t)bh * 128 * 4 * 32 * 16;
  const unsigned short* vbase = vp + (size_t)bh * 64 * 8 * 32 * 16;

  for (int kc = 0; kc < 64; ++kc) {
    v8f Ssub[2];
#pragma unroll
    for (int sub = 0; sub < 2; ++sub) {
      int jt = kc * 2 + sub;
      FragB kfr[4];                 // preload all 4 fragments, then 4 WMMAs
#pragma unroll
      for (int ks = 0; ks < 4; ++ks) {
        const unsigned short* kf =
            kbase + ((size_t)(jt * 4 + ks) * 32 + lane) * 16;
        ld2x16(kfr[ks], kf, kf + 8);
      }
      v8f s = zero8();
#pragma unroll
      for (int ks = 0; ks < 4; ++ks) s = wmma_bf16(Qf[ks], kfr[ks], s);
      Ssub[sub] = s;
    }
    // first V fragment issued early: its latency hides under softmax VALU
    FragB vfr[2];
    {
      const unsigned short* vf = vbase + (((size_t)kc * 8) * 32 + lane) * 16;
      ld2x16(vfr[0], vf, vf + 8);
    }
    // online softmax update (row stats shared across 16 n-lanes via shfl)
#pragma unroll
    for (int r = 0; r < 8; ++r) {
      float s0 = Ssub[0][r] * scale, s1 = Ssub[1][r] * scale;
      float t = fmaxf(s0, s1);
#pragma unroll
      for (int m = 1; m <= 8; m <<= 1) t = fmaxf(t, __shfl_xor(t, m, 16));
      float mnew = fmaxf(mrow[r], t);
      corr[r] = __expf(mrow[r] - mnew);
      mrow[r] = mnew;
      float p0 = __expf(s0 - mnew), p1 = __expf(s1 - mnew);
      float ps = p0 + p1;
#pragma unroll
      for (int m = 1; m <= 8; m <<= 1) ps += __shfl_xor(ps, m, 16);
      lrow[r] = lrow[r] * corr[r] + ps;
      int row = r + hf * 8;
      P[row * 40 + l16] = f2bfu(p0);       // C-layout -> LDS transpose
      P[row * 40 + 16 + l16] = f2bfu(p1);
    }
#pragma unroll
    for (int nt = 0; nt < 8; ++nt)
#pragma unroll
      for (int r = 0; r < 8; ++r) Oacc[nt][r] *= corr[r];
    __builtin_amdgcn_wave_barrier();  // LDS in-order within wave; pin ordering
    FragB pf;  // A-fragment of P (16x32)
    ld2x16(pf, P + l16 * 40 + hf * 8, P + l16 * 40 + 16 + hf * 8);
    __builtin_amdgcn_wave_barrier();
#pragma unroll
    for (int nt = 0; nt < 8; ++nt) {       // double-buffered V fragments
      int cur = nt & 1, nxt = cur ^ 1;
      if (nt < 7) {
        const unsigned short* vf =
            vbase + (((size_t)kc * 8 + nt + 1) * 32 + lane) * 16;
        ld2x16(vfr[nxt], vf, vf + 8);
      }
      Oacc[nt] = wmma_bf16(pf, vfr[cur], Oacc[nt]);
    }
  }
  // epilogue: O = qh + Oacc/l
  float inv[8];
#pragma unroll
  for (int r = 0; r < 8; ++r) inv[r] = 1.f / lrow[r];
#pragma unroll
  for (int nt = 0; nt < 8; ++nt) {
#pragma unroll
    for (int r = 0; r < 8; ++r) {
      int tr = b * NSEQ + q0 + r + hf * 8;
      int c = h * 128 + nt * 16 + l16;
      size_t idx = (size_t)tr * 512 + c;
      Xout[idx] = qf[idx] + Oacc[nt][r] * inv[r];
    }
  }
}

// ---------------------------------------------------------------------------
// Row LayerNorm over `cols`, in place; optional bf16 mirror
// ---------------------------------------------------------------------------
__global__ __launch_bounds__(256) void ln_row(float* __restrict__ X,
                                              unsigned short* Xb,
                                              const float* __restrict__ g,
                                              const float* __restrict__ bt,
                                              int cols) {
  int row = blockIdx.x, tid = threadIdx.x;
  float* xr = X + (size_t)row * cols;
  __shared__ float red[256];
  __shared__ float smean, srstd;
  float s = 0.f;
  for (int c = tid; c < cols; c += 256) s += xr[c];
  red[tid] = s; __syncthreads();
  for (int o = 128; o; o >>= 1) { if (tid < o) red[tid] += red[tid + o]; __syncthreads(); }
  if (!tid) smean = red[0] / cols;
  __syncthreads();
  float mean = smean;
  float v = 0.f;
  for (int c = tid; c < cols; c += 256) { float d = xr[c] - mean; v += d * d; }
  red[tid] = v; __syncthreads();
  for (int o = 128; o; o >>= 1) { if (tid < o) red[tid] += red[tid + o]; __syncthreads(); }
  if (!tid) srstd = rsqrtf(red[0] / cols + EPSF);
  __syncthreads();
  float rstd = srstd;
  for (int c = tid; c < cols; c += 256) {
    float y = (xr[c] - mean) * rstd * g[c] + bt[c];
    xr[c] = y;
    if (Xb) Xb[(size_t)row * cols + c] = f2bfu(y);
  }
}

// ---------------------------------------------------------------------------
// Decoder (PMA, Nq=1 per batch): small scalar kernels
// ---------------------------------------------------------------------------
__global__ __launch_bounds__(256) void qdec_kernel(const float* __restrict__ S,
                                                   const float* __restrict__ wq,
                                                   const float* __restrict__ bq,
                                                   float* __restrict__ qd) {
  int idx = blockIdx.x * 256 + threadIdx.x;
  if (idx >= 8 * 512) return;
  int o = idx & 511;
  float acc = bq[o];
  const float* wr = wq + (size_t)o * 512;
  for (int i = 0; i < 512; ++i) acc += S[i] * wr[i];
  qd[idx] = acc;
}

__global__ __launch_bounds__(256) void dec_attn(const float* __restrict__ qd,
                                                const unsigned short* __restrict__ kb,
                                                const unsigned short* __restrict__ vb,
                                                float* __restrict__ dsc,
                                                float* __restrict__ od) {
  int bh = blockIdx.x, b = bh >> 2, h = bh & 3, tid = threadIdx.x;
  const float scale = 0.044194173824159216f;
  const float* qr = qd + b * 512 + h * 128;
  __shared__ float red[256];
  __shared__ float smx, ssm;
  float lmax = -1e30f;
  for (int j = tid; j < NSEQ; j += 256) {
    const unsigned short* kr = kb + (size_t)(b * NSEQ + j) * 512 + h * 128;
    float s = 0.f;
    for (int c = 0; c < 128; ++c) s += qr[c] * bfu2f(kr[c]);
    s *= scale;
    dsc[bh * NSEQ + j] = s;
    lmax = fmaxf(lmax, s);
  }
  red[tid] = lmax; __syncthreads();
  for (int o = 128; o; o >>= 1) { if (tid < o) red[tid] = fmaxf(red[tid], red[tid + o]); __syncthreads(); }
  if (!tid) smx = red[0];
  __syncthreads();
  float mx = smx, lsum = 0.f;
  for (int j = tid; j < NSEQ; j += 256) {
    float p = __expf(dsc[bh * NSEQ + j] - mx);
    dsc[bh * NSEQ + j] = p;
    lsum += p;
  }
  __syncthreads();
  red[tid] = lsum; __syncthreads();
  for (int o = 128; o; o >>= 1) { if (tid < o) red[tid] += red[tid + o]; __syncthreads(); }
  if (!tid) ssm = red[0];
  __syncthreads();
  float inv = 1.f / ssm;
  if (tid < 128) {
    int c = tid;
    float acc = 0.f;
    for (int j = 0; j < NSEQ; ++j)
      acc += dsc[bh * NSEQ + j] * bfu2f(vb[(size_t)(b * NSEQ + j) * 512 + h * 128 + c]);
    od[b * 512 + h * 128 + c] = qr[c] + acc * inv;
  }
}

__global__ __launch_bounds__(256) void dec_ff(const float* __restrict__ H,
                                              const float* __restrict__ wo,
                                              const float* __restrict__ bo,
                                              float* __restrict__ out2) {
  int idx = blockIdx.x * 256 + threadIdx.x;
  if (idx >= 8 * 512) return;
  int r = idx >> 9, o = idx & 511;
  const float* hr = H + (size_t)r * 512;
  const float* wr = wo + (size_t)o * 512;
  float acc = bo[o];
  for (int i = 0; i < 512; ++i) acc += hr[i] * wr[i];
  out2[idx] = hr[o] + fmaxf(acc, 0.f);
}

__global__ __launch_bounds__(256) void out_fc(const float* __restrict__ H,
                                              const float* __restrict__ W,
                                              const float* __restrict__ bb,
                                              float* __restrict__ out0) {
  int idx = blockIdx.x * 256 + threadIdx.x;
  if (idx >= 8 * 256) return;
  int r = idx >> 8, o = idx & 255;
  const float* hr = H + (size_t)r * 512;
  const float* wr = W + (size_t)o * 512;
  float acc = bb[o];
  for (int i = 0; i < 512; ++i) acc += hr[i] * wr[i];
  out0[idx] = acc;
}

// BN(batch=8) -> exact GELU -> BN(batch=8); one thread per channel
__global__ __launch_bounds__(256) void final_bn(const float* __restrict__ out0,
                                                const float* g1, const float* b1,
                                                const float* g2, const float* b2,
                                                float* __restrict__ dout) {
  int o = threadIdx.x;
  if (o >= 256) return;
  float v[8];
  float m = 0.f;
#pragma unroll
  for (int r = 0; r < 8; ++r) { v[r] = out0[r * 256 + o]; m += v[r]; }
  m *= 0.125f;
  float var = 0.f;
#pragma unroll
  for (int r = 0; r < 8; ++r) { float d = v[r] - m; var += d * d; }
  float rstd = rsqrtf(var * 0.125f + EPSF);
#pragma unroll
  for (int r = 0; r < 8; ++r) {
    float y = (v[r] - m) * rstd * g1[o] + b1[o];
    v[r] = 0.5f * y * (1.f + erff(y * 0.70710678118654752f));
  }
  float m2 = 0.f;
#pragma unroll
  for (int r = 0; r < 8; ++r) m2 += v[r];
  m2 *= 0.125f;
  float var2 = 0.f;
#pragma unroll
  for (int r = 0; r < 8; ++r) { float d = v[r] - m2; var2 += d * d; }
  float rstd2 = rsqrtf(var2 * 0.125f + EPSF);
#pragma unroll
  for (int r = 0; r < 8; ++r)
    dout[r * 256 + o] = (v[r] - m2) * rstd2 * g2[o] + b2[o];
}

// ---------------------------------------------------------------------------
// host launcher
// ---------------------------------------------------------------------------
extern "C" void kernel_launch(void* const* d_in, const int* in_sizes, int n_in,
                              void* d_out, int out_size, void* d_ws, size_t ws_size,
                              hipStream_t stream) {
  (void)in_sizes; (void)n_in; (void)out_size; (void)ws_size;
#define FIN(i) ((const float*)d_in[i])

  char* ws = (char*)d_ws;
  const size_t SZ_F = (size_t)T_TOK * DHID * 4;   // 32 MB
  const size_t SZ_B = (size_t)T_TOK * DHID * 2;   // 16 MB
  const size_t WPEL = 512 * 512;                  // ushorts per packed matrix

  size_t off = 0;
  float* X = (float*)(ws + off); off += SZ_F;
  unsigned short* Xb = (unsigned short*)(ws + off); off += SZ_B;
  float* qf = (float*)(ws + off); off += SZ_F;
  unsigned short* qb = (unsigned short*)(ws + off); off += SZ_B;
  unsigned short* kb = (unsigned short*)(ws + off); off += SZ_B;
  unsigned short* vb = (unsigned short*)(ws + off); off += SZ_B;
  unsigned short* kp = (unsigned short*)(ws + off); off += SZ_B;
  unsigned short* vp = (unsigned short*)(ws + off); off += SZ_B;
  float* h5 = (float*)(ws + off); off += (size_t)T_TOK * 5 * 4;
  unsigned short* wp = (unsigned short*)(ws + off); off += 10 * WPEL * 2;
  float* stats = (float*)(ws + off); off += 256;
  float* qdec = (float*)(ws + off); off += 8 * 512 * 4;
  float* odec = (float*)(ws + off); off += 8 * 512 * 4;
  float* odec2 = (float*)(ws + off); off += 8 * 512 * 4;
  float* dsc = (float*)(ws + off); off += 32 * NSEQ * 4;
  float* out0 = (float*)(ws + off); off += 8 * 256 * 4;
  // total ~166 MB

  // 1) KAN + LN(5) + BN(5) + embedding
  zero_stats<<<1, 32, 0, stream>>>(stats);
  kan_kernel<<<64, 256, 0, stream>>>(FIN(0),
      FIN(1), FIN(2), FIN(3), FIN(4), FIN(5), FIN(6),
      FIN(7), FIN(8), FIN(9), FIN(10), FIN(11), FIN(12),
      FIN(13), FIN(14), h5);
  bn5_stats<<<64, 256, 0, stream>>>(h5, stats);
  embed_kernel<<<(T_TOK * DHID) / 256, 256, 0, stream>>>(
      h5, stats, FIN(15), FIN(16), FIN(17), FIN(18), X, Xb);

  // 2) pack weights: enc0 wq/wk/wv/wo, enc1 wq/wk/wv/wo, dec wk/wv
  const int widx[10] = {19, 21, 23, 25, 31, 33, 35, 37, 46, 48};
  for (int i = 0; i < 10; ++i)
    pack_w<<<64, 256, 0, stream>>>(FIN(widx[i]), wp + (size_t)i * WPEL);

  dim3 ggrid(T_TOK / 128, 8);

  // 3) two encoder MAB blocks
  for (int L = 0; L < 2; ++L) {
    int p = 19 + L * 12;
    unsigned short* wpq = wp + (size_t)(L * 4 + 0) * WPEL;
    unsigned short* wpk = wp + (size_t)(L * 4 + 1) * WPEL;
    unsigned short* wpv = wp + (size_t)(L * 4 + 2) * WPEL;
    unsigned short* wpo = wp + (size_t)(L * 4 + 3) * WPEL;
    gemm512<<<ggrid, 256, 0, stream>>>(Xb, wpq, FIN(p + 1), qf, qb, nullptr, 0);
    gemm512<<<ggrid, 256, 0, stream>>>(Xb, wpk, FIN(p + 3), nullptr, kb, nullptr, 0);
    gemm512<<<ggrid, 256, 0, stream>>>(Xb, wpv, FIN(p + 5), nullptr, vb, nullptr, 0);
    pack_k<<<2048, 256, 0, stream>>>(kb, kp);
    pack_v<<<2048, 256, 0, stream>>>(vb, vp);
    flash_attn<<<dim3(NSEQ / 64, 32), 128, 0, stream>>>(qb, qf, kp, vp, X);
    ln_row<<<T_TOK, 256, 0, stream>>>(X, Xb, FIN(p + 8), FIN(p + 9), 512);
    gemm512<<<ggrid, 256, 0, stream>>>(Xb, wpo, FIN(p + 7), nullptr, nullptr, X, 1);
    ln_row<<<T_TOK, 256, 0, stream>>>(X, Xb, FIN(p + 10), FIN(p + 11), 512);
  }

  // 4) decoder PMA (k/v projections over full T via WMMA; rest scalar)
  gemm512<<<ggrid, 256, 0, stream>>>(Xb, wp + 8 * WPEL, FIN(47), nullptr, kb, nullptr, 0);
  gemm512<<<ggrid, 256, 0, stream>>>(Xb, wp + 9 * WPEL, FIN(49), nullptr, vb, nullptr, 0);
  qdec_kernel<<<16, 256, 0, stream>>>(FIN(43), FIN(44), FIN(45), qdec);
  dec_attn<<<32, 256, 0, stream>>>(qdec, kb, vb, dsc, odec);
  ln_row<<<8, 256, 0, stream>>>(odec, nullptr, FIN(52), FIN(53), 512);
  dec_ff<<<16, 256, 0, stream>>>(odec, FIN(50), FIN(51), odec2);
  ln_row<<<8, 256, 0, stream>>>(odec2, nullptr, FIN(54), FIN(55), 512);

  // 5) output FC + BN + GELU + BN
  out_fc<<<8, 256, 0, stream>>>(odec2, FIN(56), FIN(57), out0);
  final_bn<<<1, 256, 0, stream>>>(out0, FIN(58), FIN(59), FIN(60), FIN(61),
                                  (float*)d_out);
#undef FIN
}
